// AutoRegressive_7078106103836
// MI455X (gfx1250) — matrix operand order repaired and checked by reading, not compiled, not run on hardware
//
#include <hip/hip_runtime.h>
#include <math.h>

// ---------------------------------------------------------------------------
// Problem constants (from reference)
// ---------------------------------------------------------------------------
#define BATCH     2048
#define T_IN      48
#define FEAT      64
#define UNITS     1024
#define OUT_STEPS 64
#define KTOT      (FEAT + UNITS)      // 1088, K of fused [x|h] @ [W;U]
#define GCOLS     (4 * UNITS)         // 4096 gate columns
#define MROWS     64                  // rows per wave in the LSTM GEMM (4 m-tiles)

typedef __bf16 bf16;
typedef bf16  v16bf __attribute__((ext_vector_type(16)));
typedef bf16  v8bf  __attribute__((ext_vector_type(8)));
typedef float v8f   __attribute__((ext_vector_type(8)));

// Build a 16-element bf16 WMMA operand from two 16-byte chunks.
__device__ __forceinline__ v16bf ld_op(const bf16* p0, const bf16* p1) {
    v8bf lo = *(const v8bf*)p0;
    v8bf hi = *(const v8bf*)p1;
    return __builtin_shufflevector(lo, hi, 0,1,2,3,4,5,6,7,8,9,10,11,12,13,14,15);
}

__device__ __forceinline__ v8f wmma_bf16(v16bf a, v16bf b, v8f c) {
    // D(f32 16x16) = A(bf16 16x32) * B(bf16 32x16) + C
    return __builtin_amdgcn_wmma_f32_16x16x32_bf16(
        /*neg_a=*/false, a, /*neg_b=*/false, b,
        /*c_mod=*/(short)0, c, /*reuse_a=*/false, /*reuse_b=*/false);
}

__device__ __forceinline__ float sigmoid_f(float x) {
    return 1.0f / (1.0f + __expf(-x));
}

// ---------------------------------------------------------------------------
// Init kernels: weight conversion (fp32 -> bf16, transposed) and state zeroing
// ---------------------------------------------------------------------------

// WUt[n][k] (n in 0..4095 gate-column, k in 0..1087) = [W;U][k][n], bf16.
__global__ void k_convert_wut(const float* __restrict__ W,
                              const float* __restrict__ U,
                              bf16* __restrict__ WUt) {
    int k = blockIdx.x * 256 + threadIdx.x;
    int n = blockIdx.y;
    if (k >= KTOT) return;
    float v = (k < FEAT) ? W[(size_t)k * GCOLS + n]
                         : U[(size_t)(k - FEAT) * GCOLS + n];
    WUt[(size_t)n * KTOT + k] = (bf16)v;
}

// Wdt[f][u] = Wd[u][f], bf16 (feat-major so B-operand reads are contiguous).
__global__ void k_convert_wdt(const float* __restrict__ Wd,
                              bf16* __restrict__ Wdt) {
    int idx = blockIdx.x * 256 + threadIdx.x;   // 64 * 1024
    if (idx >= FEAT * UNITS) return;
    int f = idx / UNITS;
    int u = idx % UNITS;
    Wdt[idx] = (bf16)Wd[(size_t)u * FEAT + f];
}

// Zero c and the h-columns of both ping-pong activation buffers.
__global__ void k_init_state(float* __restrict__ C,
                             bf16* __restrict__ A0,
                             bf16* __restrict__ A1) {
    int idx = blockIdx.x * 256 + threadIdx.x;   // BATCH * UNITS
    if (idx >= BATCH * UNITS) return;
    int row = idx >> 10;
    int u   = idx & (UNITS - 1);
    C[idx] = 0.0f;
    A0[(size_t)row * KTOT + FEAT + u] = (bf16)0.0f;
    A1[(size_t)row * KTOT + FEAT + u] = (bf16)0.0f;
}

// Stage inputs[:, t, :] into the x-columns of the current activation buffer.
__global__ void k_load_x(const float* __restrict__ in, int t,
                         bf16* __restrict__ Ax) {
    int idx = blockIdx.x * 256 + threadIdx.x;   // BATCH * FEAT
    if (idx >= BATCH * FEAT) return;
    int row = idx >> 6;
    int f   = idx & (FEAT - 1);
    Ax[(size_t)row * KTOT + f] =
        (bf16)in[(size_t)row * (T_IN * FEAT) + (size_t)t * FEAT + f];
}

// ---------------------------------------------------------------------------
// Fused LSTM step: one wave32 per (64 rows x 16 units) tile, all 4 gates.
//   z = A(bf16 [2048x1088]) @ WUt^T + b  -> gates -> new c (fp32), new h (bf16)
// Gate-major inner loop: each B operand feeds 4 m-tiles right after load,
// halving L2 weight traffic vs. a 32-row tile.
// ---------------------------------------------------------------------------
__global__ __launch_bounds__(32)
void k_lstm_step(const bf16* __restrict__ A,     // [BATCH x KTOT] (x|h), read
                 bf16* __restrict__ Hn,          // next buffer's h-part (A_next + FEAT)
                 float* __restrict__ C,          // [BATCH x UNITS] cell state (RMW)
                 const bf16* __restrict__ WUt,   // [GCOLS x KTOT] transposed weights
                 const float* __restrict__ bias) // [GCOLS]
{
    const int lane  = threadIdx.x;
    const int l15   = lane & 15;
    const int lhalf = lane >> 4;
    const int n0 = blockIdx.x * 16;      // unit block
    const int m0 = blockIdx.y * MROWS;   // row block

    v8f acc[4][4] = {};                  // [m-tile][gate]

    // A-operand: row = m0 + mt*16 + (lane&15); chunks at k0+kb and k0+16+kb.
    const bf16* ap[4];
    #pragma unroll
    for (int mt = 0; mt < 4; ++mt)
        ap[mt] = A + (size_t)(m0 + mt * 16 + l15) * KTOT;
    const int akb = lhalf * 8;

    // B-operand: col = gate*UNITS + n0 + (lane&15); chunks at k0+kb2 and k0+16+kb2.
    const bf16* bp[4];
    #pragma unroll
    for (int g = 0; g < 4; ++g)
        bp[g] = WUt + (size_t)(g * UNITS + n0 + l15) * KTOT;
    const int bkb = lhalf * 8;

    for (int k0 = 0; k0 < KTOT; k0 += 32) {
        v16bf a[4];
        #pragma unroll
        for (int mt = 0; mt < 4; ++mt)
            a[mt] = ld_op(ap[mt] + k0 + akb, ap[mt] + k0 + 16 + akb);

        #pragma unroll
        for (int g = 0; g < 4; ++g) {
            v16bf b = ld_op(bp[g] + k0 + bkb, bp[g] + k0 + 16 + bkb);
            #pragma unroll
            for (int mt = 0; mt < 4; ++mt)
                acc[mt][g] = wmma_bf16(a[mt], b, acc[mt][g]);
        }
    }

    // Epilogue: gates + cell update. Accumulator lane map:
    //   col N = n0 + (lane&15); rows M = m0 + mt*16 + 8*(lane>>4) + r.
    const int col = n0 + l15;
    const float bbi = bias[col];
    const float bbf = bias[UNITS + col];
    const float bbg = bias[2 * UNITS + col];
    const float bbo = bias[3 * UNITS + col];

    #pragma unroll
    for (int mt = 0; mt < 4; ++mt) {
        #pragma unroll
        for (int r = 0; r < 8; ++r) {
            int row = m0 + mt * 16 + lhalf * 8 + r;
            size_t cidx = (size_t)row * UNITS + col;
            float zi = acc[mt][0][r] + bbi;
            float zf = acc[mt][1][r] + bbf;
            float zg = acc[mt][2][r] + bbg;
            float zo = acc[mt][3][r] + bbo;
            float cv = C[cidx];
            float cn = sigmoid_f(zf) * cv + sigmoid_f(zi) * tanhf(zg);
            float hn = sigmoid_f(zo) * tanhf(cn);
            C[cidx] = cn;
            Hn[(size_t)row * KTOT + col] = (bf16)hn;
        }
    }
}

// ---------------------------------------------------------------------------
// Dense head: pred = h @ Wd + bd. Writes fp32 to d_out and bf16 back into the
// x-columns of the same activation buffer (input for the next decode step).
// ---------------------------------------------------------------------------
__global__ __launch_bounds__(32)
void k_dense(const bf16* __restrict__ Ah,   // A + FEAT (h part), stride KTOT
             const bf16* __restrict__ Wdt,  // [FEAT x UNITS] transposed
             const float* __restrict__ bd,  // [FEAT]
             float* __restrict__ out,       // [BATCH x OUT_STEPS x FEAT]
             bf16* __restrict__ Ax,         // A (x part), stride KTOT
             int step)
{
    const int lane  = threadIdx.x;
    const int l15   = lane & 15;
    const int lhalf = lane >> 4;
    const int n0 = blockIdx.x * 16;   // feature block (4 blocks)
    const int m0 = blockIdx.y * 32;   // row block

    v8f acc[2] = {};

    const bf16* ap0 = Ah + (size_t)(m0 + l15) * KTOT;
    const bf16* ap1 = Ah + (size_t)(m0 + 16 + l15) * KTOT;
    const int akb = lhalf * 8;

    const bf16* bp = Wdt + (size_t)(n0 + l15) * UNITS;
    const int bkb = lhalf * 8;

    for (int k0 = 0; k0 < UNITS; k0 += 32) {
        v16bf a0 = ld_op(ap0 + k0 + akb, ap0 + k0 + 16 + akb);
        v16bf a1 = ld_op(ap1 + k0 + akb, ap1 + k0 + 16 + akb);
        v16bf b  = ld_op(bp + k0 + bkb, bp + k0 + 16 + bkb);
        acc[0] = wmma_bf16(a0, b, acc[0]);
        acc[1] = wmma_bf16(a1, b, acc[1]);
    }

    const int col = n0 + l15;
    const float bb = bd[col];
    #pragma unroll
    for (int mt = 0; mt < 2; ++mt) {
        #pragma unroll
        for (int r = 0; r < 8; ++r) {
            int row = m0 + mt * 16 + lhalf * 8 + r;
            float p = acc[mt][r] + bb;
            out[(size_t)row * (OUT_STEPS * FEAT) + (size_t)step * FEAT + col] = p;
            Ax[(size_t)row * KTOT + col] = (bf16)p;
        }
    }
}

// ---------------------------------------------------------------------------
// Host orchestration
// ---------------------------------------------------------------------------
extern "C" void kernel_launch(void* const* d_in, const int* in_sizes, int n_in,
                              void* d_out, int out_size, void* d_ws, size_t ws_size,
                              hipStream_t stream) {
    const float* inputs = (const float*)d_in[0]; // [2048, 48, 64]
    const float* W      = (const float*)d_in[1]; // [64, 4096]
    const float* U      = (const float*)d_in[2]; // [1024, 4096]
    const float* b      = (const float*)d_in[3]; // [4096]
    const float* Wd     = (const float*)d_in[4]; // [1024, 64]
    const float* bd     = (const float*)d_in[5]; // [64]
    float* out          = (float*)d_out;         // [2048, 64, 64]

    // Workspace carve-up (all 256B aligned).
    char* ws = (char*)d_ws;
    size_t off = 0;
    auto carve = [&](size_t bytes) -> char* {
        char* p = ws + off;
        off = (off + bytes + 255) & ~(size_t)255;
        return p;
    };
    bf16*  A0  = (bf16*)carve((size_t)BATCH * KTOT * sizeof(bf16));    // 4.46 MB
    bf16*  A1  = (bf16*)carve((size_t)BATCH * KTOT * sizeof(bf16));    // 4.46 MB
    bf16*  WUt = (bf16*)carve((size_t)GCOLS * KTOT * sizeof(bf16));    // 8.91 MB
    bf16*  Wdt = (bf16*)carve((size_t)FEAT * UNITS * sizeof(bf16));    // 0.13 MB
    float* C   = (float*)carve((size_t)BATCH * UNITS * sizeof(float)); // 8.39 MB

    // --- init: convert weights to bf16 (transposed), zero h/c state ---
    k_convert_wut<<<dim3((KTOT + 255) / 256, GCOLS), 256, 0, stream>>>(W, U, WUt);
    k_convert_wdt<<<(FEAT * UNITS + 255) / 256, 256, 0, stream>>>(Wd, Wdt);
    k_init_state<<<(BATCH * UNITS + 255) / 256, 256, 0, stream>>>(C, A0, A1);

    const dim3 lstm_grid(UNITS / 16, BATCH / MROWS);  // 64 x 32 waves
    const dim3 dense_grid(FEAT / 16, BATCH / 32);     // 4 x 64 waves

    bf16* cur = A0;
    bf16* nxt = A1;

    // --- warmup over the input sequence ---
    for (int t = 0; t < T_IN; ++t) {
        k_load_x<<<(BATCH * FEAT + 255) / 256, 256, 0, stream>>>(inputs, t, cur);
        k_lstm_step<<<lstm_grid, 32, 0, stream>>>(cur, nxt + FEAT, C, WUt, b);
        bf16* tmp = cur; cur = nxt; nxt = tmp;        // cur now holds h_t
    }

    // --- first prediction ---
    k_dense<<<dense_grid, 32, 0, stream>>>(cur + FEAT, Wdt, bd, out, cur, 0);

    // --- autoregressive decode ---
    for (int s = 1; s < OUT_STEPS; ++s) {
        k_lstm_step<<<lstm_grid, 32, 0, stream>>>(cur, nxt + FEAT, C, WUt, b);
        bf16* tmp = cur; cur = nxt; nxt = tmp;
        k_dense<<<dense_grid, 32, 0, stream>>>(cur + FEAT, Wdt, bd, out, cur, s);
    }
}
